// MultiHeadSelfAttention_75273596829909
// MI455X (gfx1250) — matrix operand order repaired and checked by reading, not compiled, not run on hardware
//
#include <hip/hip_runtime.h>

// MI455X / gfx1250: wave32, WMMA bf16 16x16x32 fp32-acc, async LDS copies.
typedef unsigned short u16;
typedef unsigned int   u32;
typedef __attribute__((ext_vector_type(16))) __bf16 v16bf;
typedef __attribute__((ext_vector_type(8)))  float  v8f;
typedef int v4i __attribute__((vector_size(16)));

union Frag { v16bf v; uint4 u4[2]; };

static constexpr int BATCH  = 8;
static constexpr int SEQ    = 1024;
static constexpr int DMODEL = 1024;
static constexpr int HEADS  = 16;
static constexpr int DKH    = 64;

#if defined(__HIP_DEVICE_COMPILE__) && \
    __has_builtin(__builtin_amdgcn_global_load_async_to_lds_b128)
#define HAVE_ASYNC 1
#else
#define HAVE_ASYNC 0
#endif

__device__ __forceinline__ u16 f2bf(float f) {
    u32 u = __float_as_uint(f);
    u32 r = u + 0x7FFFu + ((u >> 16) & 1u);   // round-to-nearest-even
    return (u16)(r >> 16);
}

// 16-byte global -> LDS copy. Async (ASYNCcnt) on CDNA5 when available.
__device__ __forceinline__ void copy16(const u16* g, u16* l) {
#if HAVE_ASYNC
    auto gp = reinterpret_cast<__attribute__((address_space(1))) v4i*>(
        (unsigned long long)g);
    auto lp = reinterpret_cast<__attribute__((address_space(3))) v4i*>(
        (unsigned long long)l);
    __builtin_amdgcn_global_load_async_to_lds_b128(gp, lp, 0, 0);
#else
    *(uint4*)l = *(const uint4*)g;
#endif
}

__device__ __forceinline__ void wait_async() {
#if HAVE_ASYNC
#if __has_builtin(__builtin_amdgcn_s_wait_asynccnt)
    __builtin_amdgcn_s_wait_asynccnt(0);
#else
    asm volatile("s_wait_asynccnt 0" ::: "memory");
#endif
#endif
}

// ---------------------------------------------------------------------------
// fp32 -> bf16 bulk conversion (4-wide)
// ---------------------------------------------------------------------------
__global__ void cvt_f32_bf16(const float4* __restrict__ src,
                             uint2* __restrict__ dst, int n4) {
    int i = blockIdx.x * blockDim.x + threadIdx.x;
    if (i < n4) {
        float4 f = src[i];
        uint2 r;
        r.x = (u32)f2bf(f.x) | ((u32)f2bf(f.y) << 16);
        r.y = (u32)f2bf(f.z) | ((u32)f2bf(f.w) << 16);
        dst[i] = r;
    }
}

// ---------------------------------------------------------------------------
// fp32 (K x Nc) -> bf16 transposed (Nc x K). One-time transpose so the GEMM
// B-tile staging becomes straight 16B row copies (async-LDS friendly).
// ---------------------------------------------------------------------------
__global__ __launch_bounds__(256)
void cvt_transpose_bf16(const float* __restrict__ src, u16* __restrict__ dst,
                        int K, int Nc) {
    __shared__ float t[32][33];
    const int tx = threadIdx.x & 31, ty = threadIdx.x >> 5;   // 32x8
    #pragma unroll
    for (int i = 0; i < 4; ++i) {
        int r = blockIdx.y * 32 + ty + i * 8;
        t[ty + i * 8][tx] = src[(size_t)r * Nc + blockIdx.x * 32 + tx];
    }
    __syncthreads();
    #pragma unroll
    for (int i = 0; i < 4; ++i) {
        int n = blockIdx.x * 32 + ty + i * 8;
        dst[(size_t)n * K + blockIdx.y * 32 + tx] = f2bf(t[tx][ty + i * 8]);
    }
}

// ---------------------------------------------------------------------------
// bf16 GEMM: C(M x Nc) = A(M x K) @ Bt^T, Bt pre-transposed (Nc x K).
// 128x128 block tile, 8 waves (2x4), wave = 64x32 (8 C tiles), k-step 32.
// Double-buffered LDS, async global->LDS staging overlapped with WMMA.
// EPI==0: QKV epilogue -> per-head q/k bf16 (+V transposed), q scaled 1/8.
// EPI==1: fp32 out = A@B + bias.
// ---------------------------------------------------------------------------
template<int EPI>
__global__ __launch_bounds__(256)
void gemm_bf16(const u16* __restrict__ A, const u16* __restrict__ Bt,
               const float* __restrict__ bias, int M, int K, int Nc,
               u16* __restrict__ qd, u16* __restrict__ kd, u16* __restrict__ vtb,
               float* __restrict__ out) {
    (void)M;
    __shared__ u16 Al[2][128 * 40];   // [row][k] stride 40 -> 16B aligned
    __shared__ u16 Bl[2][128 * 40];   // [n][k]  stride 40

    const int tid  = threadIdx.x;
    const int wave = tid >> 5, lane = tid & 31;
    const int ln = lane & 15, hf = lane >> 4;
    const int wrow = wave >> 2, wcol = wave & 3;
    const int rowBase = blockIdx.y * 128;
    const int colBase = blockIdx.x * 128;

    v8f acc[4][2];
    #pragma unroll
    for (int a = 0; a < 4; ++a)
        #pragma unroll
        for (int b = 0; b < 2; ++b) acc[a][b] = {};

    // stage one 128x32 A tile and 128x32 Bt tile (all 16B aligned chunks)
    auto stage = [&](int buf, int ks) {
        #pragma unroll
        for (int i = 0; i < 2; ++i) {
            int id = tid + i * 256;
            int row = id >> 2, q4 = id & 3;
            copy16(A  + (size_t)(rowBase + row) * K + ks * 32 + q4 * 8,
                   &Al[buf][row * 40 + q4 * 8]);
            copy16(Bt + (size_t)(colBase + row) * K + ks * 32 + q4 * 8,
                   &Bl[buf][row * 40 + q4 * 8]);
        }
    };

    const int nk = K / 32;
    stage(0, 0);
    wait_async();
    __syncthreads();

    for (int ks = 0; ks < nk; ++ks) {
        const int cur = ks & 1;
        if (ks + 1 < nk) stage(cur ^ 1, ks + 1);   // overlap with compute

        // A fragments (16x32 bf16, ISA pair layout 2j+8h / 16+2(j-4)+8h)
        Frag af[4];
        #pragma unroll
        for (int t = 0; t < 4; ++t) {
            int m = wrow * 64 + t * 16 + ln;
            af[t].u4[0] = *(const uint4*)(&Al[cur][m * 40 + hf * 8]);
            af[t].u4[1] = *(const uint4*)(&Al[cur][m * 40 + 16 + hf * 8]);
        }
        // B fragments (32x16 bf16, ISA pair layout 2j+8h / 16+2(j-4)+8h)
        Frag bf[2];
        #pragma unroll
        for (int t = 0; t < 2; ++t) {
            int n = wcol * 32 + t * 16 + ln;
            bf[t].u4[0] = *(const uint4*)(&Bl[cur][n * 40 + hf * 8]);
            bf[t].u4[1] = *(const uint4*)(&Bl[cur][n * 40 + 16 + hf * 8]);
        }
        #pragma unroll
        for (int tr = 0; tr < 4; ++tr)
            #pragma unroll
            for (int tc = 0; tc < 2; ++tc)
                acc[tr][tc] = __builtin_amdgcn_wmma_f32_16x16x32_bf16(
                    false, af[tr].v, false, bf[tc].v, (short)0, acc[tr][tc],
                    false, false);

        wait_async();
        __syncthreads();
    }

    // epilogue: C layout row = r + 8*hf, col = ln
    #pragma unroll
    for (int tr = 0; tr < 4; ++tr) {
        #pragma unroll
        for (int tc = 0; tc < 2; ++tc) {
            int gcol = colBase + wcol * 32 + tc * 16 + ln;
            float bv = bias[gcol];
            #pragma unroll
            for (int r = 0; r < 8; ++r) {
                int grow = rowBase + wrow * 64 + tr * 16 + hf * 8 + r;
                float val = acc[tr][tc][r] + bv;
                if (EPI == 0) {
                    int which = gcol >> 10;          // 0=q 1=k 2=v
                    int dloc = gcol & 1023;
                    int hh = dloc >> 6, dk = dloc & 63;
                    int b = grow >> 10, nn = grow & 1023;
                    if (which == 0) {
                        val *= 0.125f;               // 1/sqrt(DK) folded into Q
                        qd[((size_t)(b * HEADS + hh) * SEQ + nn) * DKH + dk] =
                            f2bf(val);
                    } else if (which == 1) {
                        kd[((size_t)(b * HEADS + hh) * SEQ + nn) * DKH + dk] =
                            f2bf(val);
                    } else {
                        // V stored transposed per head: [bh][dk][n]
                        vtb[((size_t)(b * HEADS + hh) * DKH + dk) * SEQ + nn] =
                            f2bf(val);
                    }
                } else {
                    out[(size_t)grow * Nc + gcol] = val;
                }
            }
        }
    }
}

// ---------------------------------------------------------------------------
// Flash-style attention. Grid (SEQ/64, BATCH*HEADS), 128 threads (4 waves),
// wave owns 16 query rows. Scores computed transposed (keys = C rows) so K/Q
// fragments load straight from global as contiguous uint4s. V pre-transposed
// in global, staged into double-buffered LDS with async copies.
// ---------------------------------------------------------------------------
__global__ __launch_bounds__(128)
void attn_fwd(const u16* __restrict__ qg, const u16* __restrict__ kg,
              const u16* __restrict__ vtg, u16* __restrict__ og) {
    __shared__ u16 Vt[2][64 * 40];     // [dk][key], stride 40, ping-pong
    __shared__ u16 Pl[4][16 * 40];     // per-wave P: [q][key]
    __shared__ float st[4][16];        // per-wave per-query broadcast slot

    const int tid = threadIdx.x, wave = tid >> 5, lane = tid & 31;
    const int ln = lane & 15, hf = lane >> 4;
    const int bh = blockIdx.y;
    const int q0 = blockIdx.x * 64 + wave * 16;
    const u16* Q   = qg  + (size_t)bh * SEQ * DKH;
    const u16* Kp  = kg  + (size_t)bh * SEQ * DKH;
    const u16* Vtg = vtg + (size_t)bh * DKH * SEQ;

    // B fragments of Q^T (DK x 16 queries): Q rows are contiguous pairs.
    Frag bq[2];
    #pragma unroll
    for (int hd = 0; hd < 2; ++hd) {
        const u16* p = Q + (size_t)(q0 + ln) * DKH + hd * 32 + hf * 8;
        bq[hd].u4[0] = *(const uint4*)p;
        bq[hd].u4[1] = *(const uint4*)(p + 16);
    }

    // stage a 64dk x 32key V tile: 256 aligned 16B chunks
    auto stageV = [&](int buf, int kb0) {
        #pragma unroll
        for (int i = 0; i < 2; ++i) {
            int id = tid + i * 128;
            int dk = id >> 2, c = id & 3;
            copy16(Vtg + (size_t)dk * SEQ + kb0 + c * 8,
                   &Vt[buf][dk * 40 + c * 8]);
        }
    };

    v8f o[4] = {};
    float mrun = -3.0e38f, lrun = 0.f;

    stageV(0, 0);
    wait_async();
    __syncthreads();

    for (int it = 0; it < SEQ / 32; ++it) {
        const int kb0 = it * 32;
        const int cur = it & 1;
        if (it + 1 < SEQ / 32) stageV(cur ^ 1, kb0 + 32);

        // S^T tiles: rows = keys, cols = queries. 2 tiles x 2 WMMA (DK=64).
        v8f sc[2];
        #pragma unroll
        for (int kt = 0; kt < 2; ++kt) {
            v8f s = {};
            #pragma unroll
            for (int hd = 0; hd < 2; ++hd) {
                Frag a;
                const u16* p = Kp + (size_t)(kb0 + kt * 16 + ln) * DKH + hd * 32;
                a.u4[0] = *(const uint4*)(p + hf * 8);
                a.u4[1] = *(const uint4*)(p + 16 + hf * 8);
                s = __builtin_amdgcn_wmma_f32_16x16x32_bf16(
                        false, a.v, false, bq[hd].v, (short)0, s, false, false);
            }
            sc[kt] = s;
        }

        // online softmax over this 32-key block (per query column ln)
        float bm = -3.0e38f;
        #pragma unroll
        for (int kt = 0; kt < 2; ++kt)
            #pragma unroll
            for (int r = 0; r < 8; ++r) bm = fmaxf(bm, sc[kt][r]);
        bm = fmaxf(bm, __shfl_xor(bm, 16, 32));
        float mnew = fmaxf(mrun, bm);
        float corr = __expf(mrun - mnew);
        float ls = 0.f;
        #pragma unroll
        for (int kt = 0; kt < 2; ++kt)
            #pragma unroll
            for (int r = 0; r < 8; ++r) {
                float pv = __expf(sc[kt][r] - mnew);
                ls += pv;
                Pl[wave][ln * 40 + kt * 16 + hf * 8 + r] = f2bf(pv);
            }
        ls += __shfl_xor(ls, 16, 32);
        lrun = lrun * corr + ls;
        mrun = mnew;

        if (hf == 0) st[wave][ln] = corr;     // broadcast alpha per query row
        float ar[8];
        #pragma unroll
        for (int r = 0; r < 8; ++r) ar[r] = st[wave][hf * 8 + r];
        #pragma unroll
        for (int d = 0; d < 4; ++d)
            #pragma unroll
            for (int r = 0; r < 8; ++r) o[d][r] *= ar[r];

        // PV: A = P (16q x 32k) from LDS, B = V block (32k x 64dk) from Vt.
        Frag pf;
        {
            const u16* pp = &Pl[wave][ln * 40 + hf * 8];
            pf.u4[0] = *(const uint4*)pp;
            pf.u4[1] = *(const uint4*)(pp + 16);
        }
        #pragma unroll
        for (int d = 0; d < 4; ++d) {
            Frag vf;
            const u16* vp = &Vt[cur][(d * 16 + ln) * 40 + hf * 8];
            vf.u4[0] = *(const uint4*)vp;
            vf.u4[1] = *(const uint4*)(vp + 16);
            o[d] = __builtin_amdgcn_wmma_f32_16x16x32_bf16(
                       false, pf.v, false, vf.v, (short)0, o[d], false, false);
        }
        wait_async();
        __syncthreads();
    }

    // normalize by running sum, write bf16 (B, N, D) for the proj GEMM
    if (hf == 0) st[wave][ln] = lrun;
    float inv[8];
    #pragma unroll
    for (int r = 0; r < 8; ++r) inv[r] = 1.0f / st[wave][hf * 8 + r];
    const int b = bh >> 4, hh = bh & 15;
    #pragma unroll
    for (int d = 0; d < 4; ++d)
        #pragma unroll
        for (int r = 0; r < 8; ++r) {
            int qq = q0 + hf * 8 + r;
            og[(size_t)(b * SEQ + qq) * DMODEL + hh * DKH + d * 16 + ln] =
                f2bf(o[d][r] * inv[r]);
        }
}

// ---------------------------------------------------------------------------
extern "C" void kernel_launch(void* const* d_in, const int* in_sizes, int n_in,
                              void* d_out, int out_size, void* d_ws, size_t ws_size,
                              hipStream_t stream) {
    (void)in_sizes; (void)n_in; (void)out_size; (void)ws_size;
    const float* x      = (const float*)d_in[0];
    const float* w_qkv  = (const float*)d_in[1];
    const float* b_qkv  = (const float*)d_in[2];
    const float* w_proj = (const float*)d_in[3];
    const float* b_proj = (const float*)d_in[4];
    float* out = (float*)d_out;

    const size_t nX    = (size_t)BATCH * SEQ * DMODEL;       // 8388608
    const size_t nWqkv = (size_t)DMODEL * 3 * DMODEL;        // 3145728
    const size_t nWprj = (size_t)DMODEL * DMODEL;            // 1048576
    const size_t nHead = (size_t)BATCH * HEADS * SEQ * DKH;  // 8388608

    char* w = (char*)d_ws;
    u16* xb   = (u16*)w; w += nX * 2;
    u16* wqbT = (u16*)w; w += nWqkv * 2;   // transposed: (3D x D)
    u16* wpbT = (u16*)w; w += nWprj * 2;   // transposed: (D x D)
    u16* qd   = (u16*)w; w += nHead * 2;
    u16* kd   = (u16*)w; w += nHead * 2;
    u16* vtb  = (u16*)w; w += nHead * 2;   // V transposed per head
    u16* attn = (u16*)w; w += nX * 2;

    // 1) fp32 -> bf16 conversions (weights transposed once, here)
    cvt_f32_bf16<<<(int)((nX / 4 + 255) / 256), 256, 0, stream>>>(
        (const float4*)x, (uint2*)xb, (int)(nX / 4));
    cvt_transpose_bf16<<<dim3(3 * DMODEL / 32, DMODEL / 32), 256, 0, stream>>>(
        w_qkv, wqbT, DMODEL, 3 * DMODEL);
    cvt_transpose_bf16<<<dim3(DMODEL / 32, DMODEL / 32), 256, 0, stream>>>(
        w_proj, wpbT, DMODEL, DMODEL);

    // 2) QKV projection: (8192x1024) @ (1024x3072) -> q/k heads + V^T
    gemm_bf16<0><<<dim3(3 * DMODEL / 128, BATCH * SEQ / 128), 256, 0, stream>>>(
        xb, wqbT, b_qkv, BATCH * SEQ, DMODEL, 3 * DMODEL, qd, kd, vtb, nullptr);

    // 3) fused attention (flash-style, online softmax)
    attn_fwd<<<dim3(SEQ / 64, BATCH * HEADS), 128, 0, stream>>>(qd, kd, vtb, attn);

    // 4) output projection: (8192x1024) @ (1024x1024) + bias -> fp32 out
    gemm_bf16<1><<<dim3(DMODEL / 128, BATCH * SEQ / 128), 256, 0, stream>>>(
        attn, wpbT, b_proj, BATCH * SEQ, DMODEL, DMODEL,
        nullptr, nullptr, nullptr, out);
}
